// RPEMambaLayer_16527034155757
// MI455X (gfx1250) — compile-verified, matrix operand order checked by reading, not run on hardware
//
#include <hip/hip_runtime.h>
#include <math.h>

// ---------------------------------------------------------------------------
// RPEMamba layer for MI455X (gfx1250).  bf16-in / f32-accumulate WMMA GEMMs
// (software-pipelined, 32x64 wave tiles), wave32 selective scan with 16-lane
// shuffle reduction.
// ---------------------------------------------------------------------------

#define D_MODEL 256
#define D_INNER 512
#define D_STATE 16
#define DT_RANK 16
#define SEQ_N   8192
#define EPS_LN  1e-5f

typedef __attribute__((ext_vector_type(16))) __bf16 v16bf;
typedef __attribute__((ext_vector_type(8)))  float  v8f;

__device__ __forceinline__ float silu_f(float x)      { return x / (1.0f + __expf(-x)); }
__device__ __forceinline__ float softplus_f(float x)  { return (x > 20.0f) ? x : log1pf(__expf(x)); }

// ---------------------------------------------------------------------------
// WMMA fragment loader.  Works for both the A operand (row-major activations,
// lane = M row) and the B operand (weights stored [N,K] row-major -> B^T, so
// lane = N column reads a contiguous K row of W).  Layout follows the CDNA5
// 16-bit 16x32 operand spec: lanes 0-15 hold K 0..7 / 16..23 in VGPRs 0-3 /
// 4-7, lanes 16-31 hold K 8..15 / 24..31.  The scalar loads merge into
// global_load_b128 + v_cvt_pk_bf16_f32 at -O3.
// ---------------------------------------------------------------------------
__device__ __forceinline__ v16bf load_frag16x32(const float* __restrict__ base,
                                                int ld, int rc0, int k0, int lane)
{
    const int row = lane & 15;
    const int kh  = lane >> 4;
    const float* p = base + (size_t)(rc0 + row) * (size_t)ld + k0;
    v16bf f;
#pragma unroll
    for (int j = 0; j < 16; ++j) {
        const int v  = j >> 1;
        const int kk = ((v < 4) ? 0 : 16) + kh * 8 + (v & 3) * 2 + (j & 1);
        f[j] = (__bf16)p[kk];
    }
    return f;
}

// ---------------------------------------------------------------------------
// Generic GEMM: C[M,N] = act(A[M,K] @ W[N,K]^T + bias).
// Block = 256 threads (8 waves) -> 128(M) x 128(N) tile; wave = 32(M) x 64(N)
// (2 A-frags x 4 B-frags -> 8 WMMAs per k-step).  K-loop is unrolled x2 with
// ping-pong fragment buffers so the next tile's global loads issue before the
// current tile's WMMAs.  Requires N % 128 == 0, K % 64 == 0 (true here).
// ---------------------------------------------------------------------------
template <int BIAS, int ACT>
__global__ __launch_bounds__(256) void gemm_bf16_wmma(
    const float* __restrict__ A, const float* __restrict__ W,
    const float* __restrict__ bias, float* __restrict__ C,
    int K, int lda, int ldw, int ldc)
{
    const int lane = threadIdx.x & 31;
    const int wave = threadIdx.x >> 5;
    const int mBase = blockIdx.y * 128 + (wave & 3) * 32;   // 4 waves over M
    const int nBase = blockIdx.x * 128 + (wave >> 2) * 64;  // 2 waves over N

    v8f acc[2][4];
#pragma unroll
    for (int mi = 0; mi < 2; ++mi)
#pragma unroll
        for (int ni = 0; ni < 4; ++ni) acc[mi][ni] = (v8f){};

    v16bf a0[2], b0[4], a1[2], b1[4];

#define LOAD_TILE(abuf, bbuf, kk)                                             \
    do {                                                                      \
        _Pragma("unroll")                                                     \
        for (int mi = 0; mi < 2; ++mi)                                        \
            abuf[mi] = load_frag16x32(A, lda, mBase + 16 * mi, (kk), lane);   \
        _Pragma("unroll")                                                     \
        for (int ni = 0; ni < 4; ++ni)                                        \
            bbuf[ni] = load_frag16x32(W, ldw, nBase + 16 * ni, (kk), lane);   \
    } while (0)

#define MMA_TILE(abuf, bbuf)                                                  \
    do {                                                                      \
        _Pragma("unroll")                                                     \
        for (int mi = 0; mi < 2; ++mi)                                        \
            _Pragma("unroll")                                                 \
            for (int ni = 0; ni < 4; ++ni)                                    \
                acc[mi][ni] = __builtin_amdgcn_wmma_f32_16x16x32_bf16(        \
                    false, abuf[mi], false, bbuf[ni], (short)0, acc[mi][ni],  \
                    false, false);                                            \
    } while (0)

    LOAD_TILE(a0, b0, 0);
    for (int k0 = 0; k0 < K; k0 += 64) {
        if (k0 + 128 < K) {   // global_prefetch_b8, near-cache hint
            __builtin_prefetch(A + (size_t)(mBase + (lane & 15)) * lda + k0 + 128, 0, 3);
            __builtin_prefetch(W + (size_t)(nBase + (lane & 15)) * ldw + k0 + 128, 0, 3);
        }
        LOAD_TILE(a1, b1, k0 + 32);          // issue loads for k+32
        MMA_TILE(a0, b0);                    // compute k
        if (k0 + 64 < K) LOAD_TILE(a0, b0, k0 + 64);  // issue loads for k+64
        MMA_TILE(a1, b1);                    // compute k+32
    }
#undef LOAD_TILE
#undef MMA_TILE

    // C/D layout: lane holds col = lane&15; VGPR r -> row (lane>>4)*8 + r
    const int col   = lane & 15;
    const int rbase = (lane >> 4) * 8;
#pragma unroll
    for (int mi = 0; mi < 2; ++mi) {
#pragma unroll
        for (int ni = 0; ni < 4; ++ni) {
            const int mt = mBase + 16 * mi;
            const int nt = nBase + 16 * ni;
            const float bv = BIAS ? bias[nt + col] : 0.0f;
#pragma unroll
            for (int r = 0; r < 8; ++r) {
                float v = acc[mi][ni][r] + bv;
                if (ACT == 1) v = fmaxf(v, 0.0f);
                C[(size_t)(mt + rbase + r) * (size_t)ldc + nt + col] = v;
            }
        }
    }
}

// ---------------------------------------------------------------------------
// Depthwise causal conv1d (k=4) + SiLU.  xs = xz[:, 0:512] (row stride 1024).
// ---------------------------------------------------------------------------
__global__ __launch_bounds__(256) void conv_silu_kernel(
    const float* __restrict__ xz, const float* __restrict__ conv_w,
    const float* __restrict__ conv_b, float* __restrict__ xc)
{
    const int idx = blockIdx.x * blockDim.x + threadIdx.x;   // SEQ_N * D_INNER
    const int t = idx >> 9;
    const int d = idx & 511;
    float s = conv_b[d];
#pragma unroll
    for (int k = 0; k < 4; ++k) {
        const int tt = t - 3 + k;
        if (tt >= 0) s += xz[(size_t)tt * 1024 + d] * conv_w[d * 4 + k];
    }
    xc[(size_t)t * D_INNER + d] = silu_f(s);
}

// ---------------------------------------------------------------------------
// x-projection: dbl[48] = W_x @ xc[t]; dt = softplus(dbl[0:16] @ W_dt^T + b_dt)
// B = dbl[16:32], C = dbl[32:48].  One block (64 threads) per timestep.
// ---------------------------------------------------------------------------
__global__ __launch_bounds__(64) void xproj_kernel(
    const float* __restrict__ xc, const float* __restrict__ W_x,
    const float* __restrict__ W_dt, const float* __restrict__ b_dt,
    float* __restrict__ dt_out, float* __restrict__ Bm, float* __restrict__ Cm)
{
    __shared__ float xrow[D_INNER];
    __shared__ float dbl[48];
    const int t   = blockIdx.x;
    const int tid = threadIdx.x;
#pragma unroll
    for (int i = 0; i < 8; ++i) xrow[tid + 64 * i] = xc[(size_t)t * D_INNER + tid + 64 * i];
    __syncthreads();

    if (tid < 48) {
        float acc = 0.0f;
        const float* wr = W_x + tid * D_INNER;
        for (int i = 0; i < D_INNER; ++i) acc += wr[i] * xrow[i];
        dbl[tid] = acc;
    }
    __syncthreads();

#pragma unroll
    for (int d = tid; d < D_INNER; d += 64) {
        float acc = b_dt[d];
#pragma unroll
        for (int r = 0; r < DT_RANK; ++r) acc += W_dt[d * DT_RANK + r] * dbl[r];
        dt_out[(size_t)t * D_INNER + d] = softplus_f(acc);
    }
    if (tid < D_STATE)                Bm[(size_t)t * D_STATE + tid]            = dbl[16 + tid];
    else if (tid < 2 * D_STATE)       Cm[(size_t)t * D_STATE + (tid - 16)]     = dbl[32 + (tid - 16)];
}

// ---------------------------------------------------------------------------
// Selective scan.  One lane per (d, s) pair: 8192 lanes total.  Each wave
// covers 2 channels x 16 states; y = sum_s h*C via 16-lane shfl_xor reduce.
// ---------------------------------------------------------------------------
__global__ __launch_bounds__(256) void scan_kernel(
    const float* __restrict__ dt, const float* __restrict__ xc,
    const float* __restrict__ Bm, const float* __restrict__ Cm,
    const float* __restrict__ A_log, float* __restrict__ ys)
{
    const int tid = threadIdx.x;
    const int d   = blockIdx.x * 16 + (tid >> 4);   // grid 32 -> 512 channels
    const int s   = tid & 15;

    const float a = -__expf(A_log[d * D_STATE + s]);
    float h = 0.0f;

    for (int t = 0; t < SEQ_N; ++t) {
        const float dtv = dt[(size_t)t * D_INNER + d];
        const float xcv = xc[(size_t)t * D_INNER + d];
        const float bs  = Bm[(size_t)t * D_STATE + s];
        const float cs  = Cm[(size_t)t * D_STATE + s];
        h = __expf(dtv * a) * h + dtv * bs * xcv;
        float y = h * cs;
        y += __shfl_xor(y, 1);
        y += __shfl_xor(y, 2);
        y += __shfl_xor(y, 4);
        y += __shfl_xor(y, 8);
        if (s == 0) ys[(size_t)t * D_INNER + d] = y;
    }
}

// ---------------------------------------------------------------------------
// Gating: ys = (ys + Dp*xc) * silu(z),  z = xz[:, 512:1024].  In-place on ys.
// ---------------------------------------------------------------------------
__global__ __launch_bounds__(256) void gate_kernel(
    float* __restrict__ ys, const float* __restrict__ xc,
    const float* __restrict__ Dp, const float* __restrict__ xz)
{
    const int idx = blockIdx.x * blockDim.x + threadIdx.x;
    const int t = idx >> 9;
    const int d = idx & 511;
    const float z = xz[(size_t)t * 1024 + 512 + d];
    const size_t o = (size_t)t * D_INNER + d;
    ys[o] = (ys[o] + Dp[d] * xc[o]) * silu_f(z);
}

// ---------------------------------------------------------------------------
// LayerNorm(a + b) over 256 features.  One block (256 threads) per row.
// ---------------------------------------------------------------------------
__global__ __launch_bounds__(256) void ln_residual_kernel(
    const float* __restrict__ a, const float* __restrict__ b,
    const float* __restrict__ g, const float* __restrict__ beta,
    float* __restrict__ out)
{
    __shared__ float red[256];
    const int row = blockIdx.x;
    const int tid = threadIdx.x;
    const size_t o = (size_t)row * D_MODEL + tid;
    const float v = a[o] + b[o];

    red[tid] = v; __syncthreads();
#pragma unroll
    for (int sft = 128; sft > 0; sft >>= 1) { if (tid < sft) red[tid] += red[tid + sft]; __syncthreads(); }
    const float mean = red[0] * (1.0f / D_MODEL);
    __syncthreads();

    const float dv = v - mean;
    red[tid] = dv * dv; __syncthreads();
#pragma unroll
    for (int sft = 128; sft > 0; sft >>= 1) { if (tid < sft) red[tid] += red[tid + sft]; __syncthreads(); }
    const float var = red[0] * (1.0f / D_MODEL);

    out[o] = dv * rsqrtf(var + EPS_LN) * g[tid] + beta[tid];
}

// ---------------------------------------------------------------------------
extern "C" void kernel_launch(void* const* d_in, const int* in_sizes, int n_in,
                              void* d_out, int out_size, void* d_ws, size_t ws_size,
                              hipStream_t stream)
{
    const float* x      = (const float*)d_in[0];   // input_states [1,8192,256]
    // d_in[1] = position_states (unused by reference)
    const float* W_in   = (const float*)d_in[2];   // [1024,256]
    const float* conv_w = (const float*)d_in[3];   // [512,4]
    const float* conv_b = (const float*)d_in[4];   // [512]
    const float* W_x    = (const float*)d_in[5];   // [48,512]
    const float* W_dt   = (const float*)d_in[6];   // [512,16]
    const float* b_dt   = (const float*)d_in[7];   // [512]
    const float* A_log  = (const float*)d_in[8];   // [512,16]
    const float* Dp     = (const float*)d_in[9];   // [512]
    const float* W_out  = (const float*)d_in[10];  // [256,512]
    const float* W_lin  = (const float*)d_in[11];  // [256,256]
    const float* b_lin  = (const float*)d_in[12];
    const float* ln1_g  = (const float*)d_in[13];
    const float* ln1_b  = (const float*)d_in[14];
    const float* W_exp  = (const float*)d_in[15];  // [512,256]
    const float* b_exp  = (const float*)d_in[16];
    const float* W_sq   = (const float*)d_in[17];  // [256,512]
    const float* b_sq   = (const float*)d_in[18];
    const float* ln2_g  = (const float*)d_in[19];
    const float* ln2_b  = (const float*)d_in[20];

    float* ws = (float*)d_ws;
    size_t o = 0;
    float* xz   = ws + o; o += (size_t)SEQ_N * 1024;     // in-proj output
    float* xc   = ws + o; o += (size_t)SEQ_N * D_INNER;  // conv+silu (reused: ffout)
    float* dtb  = ws + o; o += (size_t)SEQ_N * D_INNER;  // dt        (reused: ffmid)
    float* Bm   = ws + o; o += (size_t)SEQ_N * D_STATE;
    float* Cm   = ws + o; o += (size_t)SEQ_N * D_STATE;
    float* ys   = ws + o; o += (size_t)SEQ_N * D_INNER;  // scan out, gated in-place
    float* mh   = ws + o; o += (size_t)SEQ_N * D_MODEL;  // mamba out-proj
    float* hlin = ws + o; o += (size_t)SEQ_N * D_MODEL;  // W_lin output
    float* h1   = ws + o; o += (size_t)SEQ_N * D_MODEL;  // LN1 output
    float* ffmid = dtb;                                   // [8192,512]
    float* ffout = xc;                                    // [8192,256]
    float* out   = (float*)d_out;                         // first 8192*256

    const dim3 blk(256);
    const int MB = SEQ_N / 128;   // 64 M-tiles

    // 1) xz = x @ W_in^T                      [8192,1024]
    gemm_bf16_wmma<0, 0><<<dim3(1024 / 128, MB), blk, 0, stream>>>(
        x, W_in, nullptr, xz, D_MODEL, D_MODEL, D_MODEL, 1024);

    // 2) depthwise causal conv + SiLU         [8192,512]
    conv_silu_kernel<<<dim3((SEQ_N * D_INNER) / 256), blk, 0, stream>>>(xz, conv_w, conv_b, xc);

    // 3) dt / B / C projections               (per-timestep block)
    xproj_kernel<<<dim3(SEQ_N), dim3(64), 0, stream>>>(xc, W_x, W_dt, b_dt, dtb, Bm, Cm);

    // 4) selective scan                       [8192,512]
    scan_kernel<<<dim3(D_INNER / 16), blk, 0, stream>>>(dtb, xc, Bm, Cm, A_log, ys);

    // 5) gate: ys = (ys + Dp*xc) * silu(z)
    gate_kernel<<<dim3((SEQ_N * D_INNER) / 256), blk, 0, stream>>>(ys, xc, Dp, xz);

    // 6) mamba out-proj: mh = ys @ W_out^T    [8192,256]
    gemm_bf16_wmma<0, 0><<<dim3(256 / 128, MB), blk, 0, stream>>>(
        ys, W_out, nullptr, mh, D_INNER, D_INNER, D_INNER, D_MODEL);

    // 7) hlin = mh @ W_lin^T + b_lin          [8192,256]
    gemm_bf16_wmma<1, 0><<<dim3(256 / 128, MB), blk, 0, stream>>>(
        mh, W_lin, b_lin, hlin, D_MODEL, D_MODEL, D_MODEL, D_MODEL);

    // 8) h1 = LN(hlin + x)
    ln_residual_kernel<<<dim3(SEQ_N), blk, 0, stream>>>(hlin, x, ln1_g, ln1_b, h1);

    // 9) ffmid = relu(h1 @ W_exp^T + b_exp)   [8192,512]
    gemm_bf16_wmma<1, 1><<<dim3(512 / 128, MB), blk, 0, stream>>>(
        h1, W_exp, b_exp, ffmid, D_MODEL, D_MODEL, D_MODEL, D_INNER);

    // 10) ffout = ffmid @ W_sq^T + b_sq       [8192,256]
    gemm_bf16_wmma<1, 0><<<dim3(256 / 128, MB), blk, 0, stream>>>(
        ffmid, W_sq, b_sq, ffout, D_INNER, D_INNER, D_INNER, D_MODEL);

    // 11) out = LN(h1 + ffout)
    ln_residual_kernel<<<dim3(SEQ_N), blk, 0, stream>>>(h1, ffout, ln2_g, ln2_b, out);

    // 12) second tuple element: input_states passthrough
    hipMemcpyAsync(out + (size_t)SEQ_N * D_MODEL, x,
                   (size_t)SEQ_N * D_MODEL * sizeof(float),
                   hipMemcpyDeviceToDevice, stream);
}
